// CifPredictorV2_86225763435373
// MI455X (gfx1250) — matrix operand order repaired and checked by reading, not compiled, not run on hardware
//
#include <hip/hip_runtime.h>
#include <hip/hip_bf16.h>
#include <math.h>

// CDNA5 / gfx1250: wave32, WMMA bf16 16x16x32 with f32 accum.
typedef __attribute__((ext_vector_type(16))) __bf16       v16bf;
typedef __attribute__((ext_vector_type(8)))  float        v8f;
typedef __attribute__((ext_vector_type(8)))  unsigned int v8u;

#define B_   16
#define T_   4096
#define T1_  4097
#define D_   512
#define TAIL 0.45f

__device__ __forceinline__ unsigned short f2bf(float f) {
  unsigned int u = __float_as_uint(f);
  u += 0x7FFFu + ((u >> 16) & 1u);          // round-to-nearest-even
  return (unsigned short)(u >> 16);
}

// ---------------------------------------------------------------------------
// Pack conv_w [co=512][ci=512][k=3] f32 -> bf16 WMMA B fragments.
// Fragment f = ((k*32 + co_tile)*16 + kc) is a 32x16 (K x N) tile, K = ci
// chunk of 32, N = co chunk of 16. Per-lane layout (wave32, 16-bit B):
//   lanes 0-15 : N = lane,    K = kc*32 + 0..7, 16..23   (16 bf16 = 32 bytes)
//   lanes 16-31: N = lane-16, K = kc*32 + 8..15, 24..31
// Stored contiguously: Wb[f*512 + lane*16 + j].
// ---------------------------------------------------------------------------
__global__ void pack_w_kernel(const float* __restrict__ conv_w,
                              unsigned short* __restrict__ Wb) {
  int f    = blockIdx.x;          // 0..1535
  int tid  = threadIdx.x;         // 0..511
  int lane = tid >> 4;
  int j    = tid & 15;
  int kc   = f & 15;
  int nt   = (f >> 4) & 31;
  int k    = f >> 9;
  int co   = nt * 16 + (lane & 15);
  int ci   = kc * 32 + ((lane >= 16) ? 8 : 0) + (j & 7) + ((j >> 3) << 4);
  float w  = conv_w[((size_t)co * D_ + ci) * 3 + k];
  Wb[(size_t)f * 512 + (size_t)lane * 16 + j] = f2bf(w);
}

// ---------------------------------------------------------------------------
// hidden f32 [B][T][D] -> bf16 [B][T+2][D] with zero pad rows (conv pad=1).
// ---------------------------------------------------------------------------
__global__ void cvt_hidden_kernel(const float* __restrict__ hidden,
                                  unsigned short* __restrict__ hb) {
  size_t i = (size_t)blockIdx.x * blockDim.x + threadIdx.x;
  size_t total = (size_t)B_ * (T_ + 2) * D_;
  if (i >= total) return;
  int d    = (int)(i % D_);
  size_t r = i / D_;
  int row  = (int)(r % (T_ + 2));
  int b    = (int)(r / (T_ + 2));
  float v = 0.f;
  if (row >= 1 && row <= T_)
    v = hidden[((size_t)b * T_ + (row - 1)) * D_ + d];
  hb[i] = f2bf(v);
}

// ---------------------------------------------------------------------------
// Fused conv(k=3) + bias + ReLU + out_w dot + sigmoid -> alphas.
// Grid: B*256 blocks (16 t-rows each), 1024 threads = 32 waves.
// Wave w owns co tile [16w, 16w+16); K loop = 3 taps x 16 chunks of 32.
// The 128MB conv activation is never written to memory.
// ---------------------------------------------------------------------------
__global__ __launch_bounds__(1024)
void conv_gemm_kernel(const unsigned short* __restrict__ hb,
                      const unsigned short* __restrict__ Wb,
                      const float* __restrict__ conv_b,
                      const float* __restrict__ out_w,
                      const float* __restrict__ out_b,
                      const float* __restrict__ mask,
                      float* __restrict__ alphas_ws,
                      float* __restrict__ alphas_out) {
  __shared__ unsigned short sA[18 * D_];  // rows t0-1 .. t0+16 (bf16)
  __shared__ float o_part[16];

  int tile = blockIdx.x;
  int b  = tile >> 8;
  int mt = tile & 255;
  int t0 = mt * 16;
  int tid = threadIdx.x;
  if (tid < 16) o_part[tid] = 0.f;

  // Stage 18 contiguous padded rows (hb row t0+r == hidden row t0+r-1).
  const unsigned int* src =
      (const unsigned int*)(hb + ((size_t)b * (T_ + 2) + t0) * D_);
  unsigned int* dst = (unsigned int*)sA;
  for (int i = tid; i < 18 * D_ / 2; i += 1024) dst[i] = src[i];
  __syncthreads();

  int w     = tid >> 5;           // wave id 0..31 -> co tile
  int lane  = tid & 31;
  int mrow  = lane & 15;          // M row this lane covers (both halves)
  int khalf = (lane >> 4) << 3;   // lanes 16-31 take K+8 sub-half

  v8f C = {0.f, 0.f, 0.f, 0.f, 0.f, 0.f, 0.f, 0.f};
  for (int k = 0; k < 3; ++k) {
    const unsigned short* srow = &sA[(size_t)(k + mrow) * D_];
    for (int kc = 0; kc < 16; ++kc) {
      int kb = kc * 32 + khalf;
      // A fragment (16x32 bf16): K = kb..kb+7 and kb+16..kb+23 (LDS, ds_load_b128)
      uint4 a0 = *(const uint4*)(srow + kb);
      uint4 a1 = *(const uint4*)(srow + kb + 16);
      // B fragment (32x16 bf16): pre-packed, 32B per lane (global_load_b128 x2)
      size_t f = (size_t)(k * 32 + w) * 16 + kc;
      const uint4* bp = (const uint4*)(Wb + f * 512 + (size_t)lane * 16);
      __builtin_prefetch((const void*)(bp + 32), 0, 1);  // 512 B ahead (half a fragment)
      uint4 b0 = bp[0];
      uint4 b1 = bp[1];
      v8u av = {a0.x, a0.y, a0.z, a0.w, a1.x, a1.y, a1.z, a1.w};
      v8u bv = {b0.x, b0.y, b0.z, b0.w, b1.x, b1.y, b1.z, b1.w};
      C = __builtin_amdgcn_wmma_f32_16x16x32_bf16(
              false, __builtin_bit_cast(v16bf, av),
              false, __builtin_bit_cast(v16bf, bv),
              (short)0, C, false, false);
    }
  }

  // Epilogue: C[r] = y[t0+m][co] with m = r + (lane>=16 ? 8 : 0), co = w*16+(lane&15).
  int co  = (w << 4) + (lane & 15);
  float cb = conv_b[co];
  float ow = out_w[co];
  int mhi = (lane >> 4) << 3;
#pragma unroll
  for (int r = 0; r < 8; ++r) {
    float v = C[r] + cb;
    v = v > 0.f ? v : 0.f;          // ReLU
    float p = v * ow;               // contribution to o[t]
    p += __shfl_xor(p, 1, 16);      // reduce over the 16 co lanes of this half
    p += __shfl_xor(p, 2, 16);
    p += __shfl_xor(p, 4, 16);
    p += __shfl_xor(p, 8, 16);
    if ((lane & 15) == 0) atomicAdd(&o_part[r + mhi], p);  // ds_add_f32
  }
  __syncthreads();

  if (tid < 16) {
    int t   = t0 + tid;
    float o = o_part[tid] + out_b[0];
    float a = 1.f / (1.f + __expf(-o));      // sigmoid; relu(sig*1-0)==sig
    a *= mask[(size_t)b * T_ + t];
    alphas_ws[(size_t)b * T1_ + t]  = a;
    alphas_out[(size_t)b * T1_ + t] = a;
  }
}

// ---------------------------------------------------------------------------
// Per-batch serial CIF scan: cumsum, fire detection, fractional remainders.
// ---------------------------------------------------------------------------
__global__ void scan_kernel(float* __restrict__ alphas_ws,
                            float* __restrict__ alphas_out,
                            int* __restrict__ fireT,
                            float* __restrict__ fireFrac,
                            int* __restrict__ nf,
                            float* __restrict__ token_num) {
  int b = threadIdx.x;
  if (b >= B_) return;
  alphas_ws[(size_t)b * T1_ + T_]  = TAIL;   // tail element (mask is ones)
  alphas_out[(size_t)b * T1_ + T_] = TAIL;
  float ps = 0.f, prev_psf = 0.f;
  int k = 0;
  for (int t = 0; t < T1_; ++t) {
    ps += alphas_ws[(size_t)b * T1_ + t];
    float psf = floorf(ps);
    if (psf - prev_psf > 0.f) {
      fireT[(size_t)b * T1_ + k]    = t;
      fireFrac[(size_t)b * T1_ + k] = ps - psf;
      ++k;
    }
    prev_psf = psf;
  }
  nf[b] = k;
  token_num[b] = floorf(ps);
}

// ---------------------------------------------------------------------------
// frame_k = sum_{t in (t_{k-1}, t_k]} alpha[t]*h[t]
//           - frac_k * h[t_k] + frac_{k-1} * h[t_{k-1}]   (h[T] == 0)
// One block per (frame k, batch b); 256 threads x 2 channels.
// ---------------------------------------------------------------------------
__global__ __launch_bounds__(256)
void cif_kernel(const float* __restrict__ hidden,
                const float* __restrict__ alphas_ws,
                const int* __restrict__ fireT,
                const float* __restrict__ fireFrac,
                const int* __restrict__ nf,
                float* __restrict__ acoustic, int Ntok) {
  int kk = blockIdx.x, b = blockIdx.y;
  int d = threadIdx.x;
  float acc0 = 0.f, acc1 = 0.f;
  if (kk < nf[b]) {
    int t1 = fireT[(size_t)b * T1_ + kk];
    int t0 = kk ? fireT[(size_t)b * T1_ + kk - 1] + 1 : 0;
    const float* Ab = alphas_ws + (size_t)b * T1_;
    for (int t = t0; t <= t1; ++t) {
      if (t < T_) {
        float a = Ab[t];
        const float* hr = hidden + ((size_t)b * T_ + t) * D_;
        acc0 += a * hr[d];
        acc1 += a * hr[d + 256];
      }
    }
    float fr1 = fireFrac[(size_t)b * T1_ + kk];
    if (t1 < T_) {
      const float* hr = hidden + ((size_t)b * T_ + t1) * D_;
      acc0 -= fr1 * hr[d];
      acc1 -= fr1 * hr[d + 256];
    }
    if (kk) {
      int tp = fireT[(size_t)b * T1_ + kk - 1];
      if (tp < T_) {
        float fr0 = fireFrac[(size_t)b * T1_ + kk - 1];
        const float* hr = hidden + ((size_t)b * T_ + tp) * D_;
        acc0 += fr0 * hr[d];
        acc1 += fr0 * hr[d + 256];
      }
    }
  }
  float* out = acoustic + ((size_t)b * Ntok + kk) * D_;
  out[d] = acc0;          // frames past nf[b] are zeroed (reference `valid`)
  out[d + 256] = acc1;
}

// ---------------------------------------------------------------------------
extern "C" void kernel_launch(void* const* d_in, const int* in_sizes, int n_in,
                              void* d_out, int out_size, void* d_ws, size_t ws_size,
                              hipStream_t stream) {
  const float* hidden = (const float*)d_in[0];
  const float* mask   = (const float*)d_in[1];
  const float* conv_w = (const float*)d_in[2];
  const float* conv_b = (const float*)d_in[3];
  const float* out_w  = (const float*)d_in[4];
  const float* out_b  = (const float*)d_in[5];
  (void)in_sizes; (void)n_in; (void)ws_size;

  // Recover data-dependent frame count from the output size:
  // out = acoustic[B,Ntok,D] ++ token_num[B] ++ alphas[B,T+1]
  long long nt = ((long long)out_size - B_ - (long long)B_ * T1_) /
                 ((long long)B_ * D_);
  int Ntok = nt > 0 ? (int)nt : 0;

  char* ws = (char*)d_ws;
  unsigned short* Wb       = (unsigned short*)(ws);             // 1,572,864 B
  unsigned short* hb       = (unsigned short*)(ws + 1572864);   // 67,141,632 B
  float*          alphasW  = (float*)(ws + 68714496);           // 262,208 B
  int*            fireT    = (int*)  (ws + 68976704);           // 262,208 B
  float*          fireFrac = (float*)(ws + 69238912);           // 262,208 B
  int*            nf       = (int*)  (ws + 69501120);           // 64 B

  float* out        = (float*)d_out;
  float* acoustic   = out;
  float* token_num  = out + (size_t)B_ * Ntok * D_;
  float* alphas_out = token_num + B_;

  hipLaunchKernelGGL(pack_w_kernel, dim3(1536), dim3(512), 0, stream, conv_w, Wb);

  size_t total = (size_t)B_ * (T_ + 2) * D_;
  hipLaunchKernelGGL(cvt_hidden_kernel, dim3((unsigned)((total + 255) / 256)),
                     dim3(256), 0, stream, hidden, hb);

  hipLaunchKernelGGL(conv_gemm_kernel, dim3(B_ * 256), dim3(1024), 0, stream,
                     hb, Wb, conv_b, out_w, out_b, mask, alphasW, alphas_out);

  hipLaunchKernelGGL(scan_kernel, dim3(1), dim3(16), 0, stream,
                     alphasW, alphas_out, fireT, fireFrac, nf, token_num);

  if (Ntok > 0)
    hipLaunchKernelGGL(cif_kernel, dim3(Ntok, B_), dim3(256), 0, stream,
                       hidden, alphasW, fireT, fireFrac, nf, acoustic, Ntok);
}
